// TransformerEncoderBlock_44839458570847
// MI455X (gfx1250) — matrix operand order repaired and checked by reading, not compiled, not run on hardware
//
#include <hip/hip_runtime.h>
#include <hip/hip_bf16.h>

typedef __attribute__((ext_vector_type(16))) _Float16 v16h;
typedef __attribute__((ext_vector_type(8)))  _Float16 v8h;
typedef __attribute__((ext_vector_type(8)))  float    v8f;
typedef __attribute__((ext_vector_type(4)))  float    v4f;

#define CB 2
#define CS 2048
#define CH 1024
#define CNH 16
#define CDK 64
#define CFF 4096

// ---------------------------------------------------------------------------
// Transpose + f32->f16 convert:  W[K][N] (row-major f32)  ->  Wt[N][K] f16
// ---------------------------------------------------------------------------
__global__ __launch_bounds__(256) void tcvt_kernel(const float* __restrict__ W,
                                                   _Float16* __restrict__ Wt,
                                                   int K, int N) {
  long idx = (long)blockIdx.x * 256 + threadIdx.x;
  long total = (long)K * N;
  if (idx < total) {
    long k = idx / N;
    long n = idx - k * N;
    Wt[n * (long)K + k] = (_Float16)W[idx];
  }
}

// ---------------------------------------------------------------------------
// LayerNorm over last dim (H=1024), one row per block, f32 in -> f16 out
// ---------------------------------------------------------------------------
__global__ __launch_bounds__(256) void ln_f16_kernel(const float* __restrict__ x,
                                                     const float* __restrict__ g,
                                                     const float* __restrict__ b,
                                                     _Float16* __restrict__ out) {
  __shared__ float red[256];
  const int tid = threadIdx.x;
  const long row = blockIdx.x;
  const float* xr = x + row * (long)CH;

  float v[4];
  float s = 0.f;
#pragma unroll
  for (int i = 0; i < 4; ++i) { v[i] = xr[tid + i * 256]; s += v[i]; }
  red[tid] = s; __syncthreads();
  for (int off = 128; off > 0; off >>= 1) {
    if (tid < off) red[tid] += red[tid + off];
    __syncthreads();
  }
  const float mu = red[0] * (1.f / (float)CH);
  __syncthreads();

  float s2 = 0.f;
#pragma unroll
  for (int i = 0; i < 4; ++i) { float d = v[i] - mu; s2 += d * d; }
  red[tid] = s2; __syncthreads();
  for (int off = 128; off > 0; off >>= 1) {
    if (tid < off) red[tid] += red[tid + off];
    __syncthreads();
  }
  const float rs = rsqrtf(red[0] * (1.f / (float)CH) + 1e-5f);

#pragma unroll
  for (int i = 0; i < 4; ++i) {
    int c = tid + i * 256;
    out[row * (long)CH + c] = (_Float16)((v[i] - mu) * rs * g[c] + b[c]);
  }
}

// ---------------------------------------------------------------------------
// Row softmax in place, S=2048 elements per row, one row per block
// ---------------------------------------------------------------------------
__global__ __launch_bounds__(256) void softmax_kernel(float* __restrict__ att) {
  __shared__ float red[256];
  const int tid = threadIdx.x;
  float* row = att + (size_t)blockIdx.x * CS;

  float v[8];
  float m = -3.0e38f;
#pragma unroll
  for (int i = 0; i < 8; ++i) { v[i] = row[tid + i * 256]; m = fmaxf(m, v[i]); }
  red[tid] = m; __syncthreads();
  for (int off = 128; off > 0; off >>= 1) {
    if (tid < off) red[tid] = fmaxf(red[tid], red[tid + off]);
    __syncthreads();
  }
  m = red[0]; __syncthreads();

  float s = 0.f;
#pragma unroll
  for (int i = 0; i < 8; ++i) { v[i] = __expf(v[i] - m); s += v[i]; }
  red[tid] = s; __syncthreads();
  for (int off = 128; off > 0; off >>= 1) {
    if (tid < off) red[tid] += red[tid + off];
    __syncthreads();
  }
  const float inv = 1.f / red[0];
#pragma unroll
  for (int i = 0; i < 8; ++i) row[tid + i * 256] = v[i] * inv;
}

// ---------------------------------------------------------------------------
// V [token][h*64+d] f16  ->  Vt [b][h][d][s] f16   (B-layout for att@V)
// ---------------------------------------------------------------------------
__global__ __launch_bounds__(256) void vtr_kernel(const _Float16* __restrict__ V,
                                                  _Float16* __restrict__ Vt) {
  long idx = (long)blockIdx.x * 256 + threadIdx.x;   // < 4096*1024
  int t = (int)(idx >> 10);       // global token
  int c = (int)(idx & 1023);
  int b = t >> 11, s = t & 2047;
  int h = c >> 6,  d = c & 63;
  Vt[(((long)(b * CNH + h) * CDK + d) << 11) + s] = V[idx];
}

// ---------------------------------------------------------------------------
// Batched WMMA GEMM:  C[M,N] = act( alpha * A[M,K] @ Bt[N,K]^T + bias ) + resid
// Block = 256 thr = 8 waves, block tile 128x128; wave tile 32(M) x 64(N).
// All epilogue modes are compile-time template flags -> straight-line epilogue.
// M is always a multiple of 128 (no row guard); GUARD_N only for N=64 (att@V).
// ---------------------------------------------------------------------------
__device__ inline float gelu_exact(float v) {
  return 0.5f * v * (1.f + erff(v * 0.70710678118654752f));
}

template <int A_F32, int HAS_BIAS, int HAS_RESID, int HAS_OUTF, int HAS_OUTH,
          int ACT, int GUARD_N>
__global__ __launch_bounds__(256) void wmma_gemm_kernel(
    const void* __restrict__ Aall, const _Float16* __restrict__ Btall,
    const float* __restrict__ bias, const float* __restrict__ residAll,
    float* __restrict__ outF, _Float16* __restrict__ outH,
    int M, int N, int K, int lda, int ldb, int ldo,
    long sA_b, long sA_h, long sB_b, long sB_h,
    long sO_b, long sO_h, long sR_b, long sR_h,
    int nh, float alpha) {
  const int z  = blockIdx.z;
  const int bb = z / nh, hh = z - bb * nh;
  const long offA = bb * sA_b + hh * sA_h;
  const long offB = bb * sB_b + hh * sB_h;
  const long offO = bb * sO_b + hh * sO_h;
  const long offR = bb * sR_b + hh * sR_h;

  const int lane = threadIdx.x & 31;
  const int wave = threadIdx.x >> 5;
  const int wm = wave & 3;          // M sub-tile: 32 rows each
  const int wn = wave >> 2;         // N sub-tile: 64 cols each
  const int hi = lane >> 4;         // lane >= 16
  const int ln = lane & 15;

  const int blockM = blockIdx.y * 128;
  const int blockN = blockIdx.x * 128;

  const _Float16* Ah = (const _Float16*)Aall;
  const float*    Af = (const float*)Aall;

  v8f acc[2][4];
#pragma unroll
  for (int t = 0; t < 2; ++t)
#pragma unroll
    for (int u = 0; u < 4; ++u)
#pragma unroll
      for (int i = 0; i < 8; ++i) acc[t][u][i] = 0.f;

  for (int kk = 0; kk < K; kk += 32) {
    v16h afrag[2], bfrag[4];
#pragma unroll
    for (int t = 0; t < 2; ++t) {
      const long row  = blockM + wm * 32 + t * 16 + ln;
      const long base = offA + row * (long)lda + kk + (hi ? 8 : 0);
      if (A_F32) {
        const float* ap = Af + base;
        v4f l0 = *(const v4f*)(ap);
        v4f l1 = *(const v4f*)(ap + 4);
        v4f h0 = *(const v4f*)(ap + 16);
        v4f h1 = *(const v4f*)(ap + 20);
#pragma unroll
        for (int i = 0; i < 4; ++i) {
          afrag[t][i]      = (_Float16)l0[i];
          afrag[t][i + 4]  = (_Float16)l1[i];
          afrag[t][i + 8]  = (_Float16)h0[i];
          afrag[t][i + 12] = (_Float16)h1[i];
        }
      } else {
        v8h lo = *(const v8h*)(Ah + base);
        v8h hb = *(const v8h*)(Ah + base + 16);
#pragma unroll
        for (int i = 0; i < 8; ++i) { afrag[t][i] = lo[i]; afrag[t][i + 8] = hb[i]; }
      }
    }
#pragma unroll
    for (int u = 0; u < 4; ++u) {
      const long col = blockN + wn * 64 + u * 16 + ln;
      const _Float16* bp = Btall + offB + col * (long)ldb + kk + hi * 8;
      v8h blo = *(const v8h*)(bp);
      v8h bhi = *(const v8h*)(bp + 16);
#pragma unroll
      for (int i = 0; i < 8; ++i) { bfrag[u][i] = blo[i]; bfrag[u][i + 8] = bhi[i]; }
    }
#pragma unroll
    for (int t = 0; t < 2; ++t)
#pragma unroll
      for (int u = 0; u < 4; ++u)
        acc[t][u] = __builtin_amdgcn_wmma_f32_16x16x32_f16(
            false, afrag[t], false, bfrag[u], (short)0, acc[t][u], false, false);
  }

  // Epilogue: straight-line (all flags compile-time)
#pragma unroll
  for (int t = 0; t < 2; ++t) {
#pragma unroll
    for (int u = 0; u < 4; ++u) {
      const int col = blockN + wn * 64 + u * 16 + ln;
      if (!GUARD_N || col < N) {
        const float bval = HAS_BIAS ? bias[col] : 0.f;
        const int row0 = blockM + wm * 32 + t * 16 + hi * 8;
#pragma unroll
        for (int i = 0; i < 8; ++i) {
          float v = acc[t][u][i] * alpha + bval;
          if (ACT) v = gelu_exact(v);
          const long o = offO + (long)(row0 + i) * ldo + col;
          if (HAS_RESID) v += residAll[offR + (long)(row0 + i) * ldo + col];
          if (HAS_OUTF) outF[o] = v;
          if (HAS_OUTH) outH[o] = (_Float16)v;
        }
      }
    }
  }
}

// ---------------------------------------------------------------------------
extern "C" void kernel_launch(void* const* d_in, const int* in_sizes, int n_in,
                              void* d_out, int out_size, void* d_ws, size_t ws_size,
                              hipStream_t stream) {
  const float* x    = (const float*)d_in[0];
  const float* ln_g = (const float*)d_in[1];
  const float* ln_b = (const float*)d_in[2];
  const float* wq   = (const float*)d_in[3];
  const float* bq   = (const float*)d_in[4];
  const float* wk   = (const float*)d_in[5];
  const float* bk   = (const float*)d_in[6];
  const float* wv   = (const float*)d_in[7];
  const float* bv   = (const float*)d_in[8];
  const float* w1   = (const float*)d_in[9];
  const float* b1   = (const float*)d_in[10];
  const float* w2   = (const float*)d_in[11];
  const float* b2   = (const float*)d_in[12];

  const int MT = CB * CS;                       // 4096 tokens
  const long SS = (long)CS * CS;

  float* xout = (float*)d_out;                  // [4096,1024]  (also x1 scratch)
  float* att  = xout + (size_t)MT * CH;         // [2,16,2048,2048]

  // ---- workspace layout (f16 elements) ----
  _Float16* h16  = (_Float16*)d_ws;             // 4096x1024 (LN1, later LN2)
  _Float16* q16  = h16 + (size_t)MT * CH;       // 4096x1024
  _Float16* k16  = q16 + (size_t)MT * CH;
  _Float16* v16  = k16 + (size_t)MT * CH;
  _Float16* vt16 = v16 + (size_t)MT * CH;       // [2,16,64,2048] (+64-row pad)
  _Float16* g16  = q16;                         // gelu 4096x4096 reuses q/k/v/vt
  _Float16* wqT  = vt16 + (size_t)MT * CH + (size_t)CDK * CS;  // after vt pad
  _Float16* wkT  = wqT + (size_t)CH * CH;
  _Float16* wvT  = wkT + (size_t)CH * CH;
  _Float16* w1T  = wvT + (size_t)CH * CH;       // [4096][1024]
  _Float16* w2T  = w1T + (size_t)CFF * CH;      // [1024][4096]

  // ---- 1. weight transpose/convert ----
  {
    int n;
    n = CH * CH;  tcvt_kernel<<<(n + 255) / 256, 256, 0, stream>>>(wq, wqT, CH, CH);
    tcvt_kernel<<<(n + 255) / 256, 256, 0, stream>>>(wk, wkT, CH, CH);
    tcvt_kernel<<<(n + 255) / 256, 256, 0, stream>>>(wv, wvT, CH, CH);
    n = CH * CFF; tcvt_kernel<<<(n + 255) / 256, 256, 0, stream>>>(w1, w1T, CH, CFF);
    tcvt_kernel<<<(n + 255) / 256, 256, 0, stream>>>(w2, w2T, CFF, CH);
  }

  // ---- 2. LN1: x -> h16 ----
  ln_f16_kernel<<<MT, 256, 0, stream>>>(x, ln_g, ln_b, h16);

  // ---- 3. QKV GEMMs: [4096,1024] = h16 @ w*T  (f16 out) ----
  {
    dim3 g(CH / 128, MT / 128, 1);
    wmma_gemm_kernel<0,1,0,0,1,0,0><<<g, 256, 0, stream>>>(h16, wqT, bq, nullptr,
        nullptr, q16, MT, CH, CH, CH, CH, CH, 0, 0, 0, 0, 0, 0, 0, 0, 1, 1.0f);
    wmma_gemm_kernel<0,1,0,0,1,0,0><<<g, 256, 0, stream>>>(h16, wkT, bk, nullptr,
        nullptr, k16, MT, CH, CH, CH, CH, CH, 0, 0, 0, 0, 0, 0, 0, 0, 1, 1.0f);
    wmma_gemm_kernel<0,1,0,0,1,0,0><<<g, 256, 0, stream>>>(h16, wvT, bv, nullptr,
        nullptr, v16, MT, CH, CH, CH, CH, CH, 0, 0, 0, 0, 0, 0, 0, 0, 1, 1.0f);
  }

  // ---- 4. V head-transpose for att@V B-operand ----
  vtr_kernel<<<(MT * CH) / 256, 256, 0, stream>>>(v16, vt16);

  // ---- 5. scores = Q @ K^T / 8  -> att (f32, in d_out) ----
  {
    dim3 g(CS / 128, CS / 128, CB * CNH);
    wmma_gemm_kernel<0,0,0,1,0,0,0><<<g, 256, 0, stream>>>(q16, k16, nullptr,
        nullptr, att, nullptr,
        CS, CS, CDK, CH, CH, CS,
        (long)CS * CH, CDK, (long)CS * CH, CDK,
        (long)CNH * SS, SS, 0, 0, CNH, 0.125f);
  }

  // ---- 6. softmax rows in place ----
  softmax_kernel<<<CB * CNH * CS, 256, 0, stream>>>(att);

  // ---- 7. att_out = att @ V ; x1 = x + att_out  -> xout (d_out) ----
  {
    dim3 g(1, CS / 128, CB * CNH);
    wmma_gemm_kernel<1,0,1,1,0,0,1><<<g, 256, 0, stream>>>(att, vt16, nullptr,
        x, xout, nullptr,
        CS, CDK, CS, CS, CS, CH,
        (long)CNH * SS, SS, (long)CNH * CDK * CS, (long)CDK * CS,
        (long)CS * CH, CDK, (long)CS * CH, CDK, CNH, 1.0f);
  }

  // ---- 8. LN2: x1 -> h16 (reuse) ----
  ln_f16_kernel<<<MT, 256, 0, stream>>>(xout, ln_g, ln_b, h16);

  // ---- 9. MLP1: gelu(h16 @ w1T + b1) -> g16 f16 ----
  {
    dim3 g(CFF / 128, MT / 128, 1);
    wmma_gemm_kernel<0,1,0,0,1,1,0><<<g, 256, 0, stream>>>(h16, w1T, b1, nullptr,
        nullptr, g16, MT, CFF, CH, CH, CH, CFF, 0, 0, 0, 0, 0, 0, 0, 0, 1, 1.0f);
  }

  // ---- 10. MLP2: xout = x1 + (g16 @ w2T + b2)   (in-place residual) ----
  {
    dim3 g(CH / 128, MT / 128, 1);
    wmma_gemm_kernel<0,1,1,1,0,0,0><<<g, 256, 0, stream>>>(g16, w2T, b2, xout,
        xout, nullptr, MT, CH, CFF, CFF, CFF, CH, 0, 0, 0, 0, 0, 0, 0, 0, 1, 1.0f);
  }
}
